// GraphGPSLayer_53463752900655
// MI455X (gfx1250) — matrix operand order repaired and checked by reading, not compiled, not run on hardware
//
#include <hip/hip_runtime.h>

// GraphGPS layer: GATv2 + MHA + FFN with LayerNorms, for gfx1250 (MI455X).
// Dense math in bf16 WMMA (v_wmma_f32_16x16x32_bf16), f32 accumulate.
// N=4096, C=512, H=8, CO=64.

#define NN 4096
#define CC 512
#define HH 8
#define COH 64

typedef __attribute__((ext_vector_type(16))) __bf16 bf16x16;
typedef __attribute__((ext_vector_type(8)))  float  f32x8;
typedef __attribute__((ext_vector_type(8)))  unsigned short u16x8;
typedef unsigned short u16;

union FragAB { bf16x16 f; u16x8 h[2]; };

__device__ __forceinline__ u16 f2bf(float x) {
  unsigned u = __float_as_uint(x);
  u += 0x7FFFu + ((u >> 16) & 1u);            // round-to-nearest-even
  return (u16)(u >> 16);
}

__device__ __forceinline__ f32x8 wmma_bf16(bf16x16 a, bf16x16 b, f32x8 c) {
  // (neg_a, A, neg_b, B, c_mod, C, reuse_a, reuse_b)
  return __builtin_amdgcn_wmma_f32_16x16x32_bf16(false, a, false, b, (short)0, c,
                                                 false, false);
}

// A-matrix fragment (16x32 bf16, MxK). ISA layout: row M = lane&15 for both
// half-waves; lane half h holds K = {8h..8h+7} in v0-3 and {16+8h..23+8h} in v4-7.
__device__ __forceinline__ bf16x16 load_a_frag(const u16* __restrict__ X, int ldx,
                                               int m0, int k0, int lane) {
  int half = lane >> 4, r = lane & 15;
  const u16* p = X + (size_t)(m0 + r) * ldx + k0 + 8 * half;
  FragAB fr;
  fr.h[0] = *(const u16x8*)(p);        // K = k0+8h .. +7
  fr.h[1] = *(const u16x8*)(p + 16);   // K = k0+16+8h .. +7
  return fr.f;
}

// B-matrix fragment (32x16 bf16, KxN), source stored as W[n, k] row-major
// (i.e. B[k,n] = W[n,k]). Column N = lane&15, K = {8h..8h+7} then {16+8h..23+8h}.
__device__ __forceinline__ bf16x16 load_b_frag(const u16* __restrict__ W, int ldw,
                                               int n0, int k0, int lane) {
  int half = lane >> 4, c = lane & 15;
  const u16* p = W + (size_t)(n0 + c) * ldw + k0 + 8 * half;
  FragAB fr;
  fr.h[0] = *(const u16x8*)(p);
  fr.h[1] = *(const u16x8*)(p + 16);
  return fr.f;
}

// ---------------------------------------------------------------- utilities
__global__ void cast_f32_to_bf16(const float* __restrict__ s, u16* __restrict__ d, int n) {
  for (int i = blockIdx.x * blockDim.x + threadIdx.x; i < n; i += gridDim.x * blockDim.x)
    d[i] = f2bf(s[i]);
}

__global__ void fill_zero_f32(float* __restrict__ p, int n) {
  for (int i = blockIdx.x * blockDim.x + threadIdx.x; i < n; i += gridDim.x * blockDim.x)
    p[i] = 0.0f;
}

// ------------------------------------------------------------------- GEMM
// Y[m, n] = sum_k X[m,k] * W[n,k] (+ bias[n]) (+ GELU), Y as f32 or bf16.
// Block = 256 threads = 8 waves arranged 2(M)x4(N); wave tile = 32x64:
// two A fragments share four B fragments -> 8 WMMAs per 6KB of L2 traffic.
template <bool WRITE_BF16, bool HAS_BIAS, bool GELU_ACT>
__global__ __launch_bounds__(256)
void gemm_bf16_kernel(const u16* __restrict__ X, const u16* __restrict__ W,
                      const float* __restrict__ bias,
                      float* __restrict__ Yf, u16* __restrict__ Yb,
                      int Nout, int K) {
  int wave = threadIdx.x >> 5, lane = threadIdx.x & 31;
  int half = lane >> 4, l15 = lane & 15;
  int m0 = blockIdx.y * 64 + (wave >> 2) * 32;         // 32 rows per wave
  int n0 = blockIdx.x * 256 + (wave & 3) * 64;         // 64 cols per wave

  f32x8 acc[2][4];
#pragma unroll
  for (int r = 0; r < 2; ++r)
#pragma unroll
    for (int t = 0; t < 4; ++t) acc[r][t] = f32x8{};

  for (int k0 = 0; k0 < K; k0 += 32) {
    bf16x16 a0 = load_a_frag(X, K, m0,      k0, lane);
    bf16x16 a1 = load_a_frag(X, K, m0 + 16, k0, lane);
#pragma unroll
    for (int t = 0; t < 4; ++t) {
      bf16x16 b = load_b_frag(W, K, n0 + 16 * t, k0, lane);
      acc[0][t] = wmma_bf16(a0, b, acc[0][t]);
      acc[1][t] = wmma_bf16(a1, b, acc[1][t]);
    }
  }
#pragma unroll
  for (int r = 0; r < 2; ++r) {
#pragma unroll
    for (int t = 0; t < 4; ++t) {
      int col = n0 + 16 * t + l15;
      float bv = HAS_BIAS ? bias[col] : 0.0f;
#pragma unroll
      for (int v = 0; v < 8; ++v) {
        int row = m0 + 16 * r + v + 8 * half;
        float val = acc[r][t][v] + bv;
        if (GELU_ACT) val = 0.5f * val * (1.0f + erff(val * 0.70710678118654752f));
        if (WRITE_BF16) Yb[(size_t)row * Nout + col] = f2bf(val);
        else            Yf[(size_t)row * Nout + col] = val;
      }
    }
  }
}

// ------------------------------------------------------------- GATv2 edges
// One thread per (edge, head). ealpha = exp(att . leakyrelu(xl[dst]+xr[src])).
// Softmax max-subtraction dropped (mathematically identical ratio; scores tiny).
__global__ void gat_edge_kernel(const float* __restrict__ xl, const float* __restrict__ xr,
                                const int* __restrict__ ei, const float* __restrict__ att,
                                float* __restrict__ ealpha, float* __restrict__ denom,
                                int E) {
  int t = blockIdx.x * blockDim.x + threadIdx.x;
  int EN = E + NN;
  if (t >= EN * HH) return;
  int e = t >> 3, h = t & 7;
  int s, d;
  if (e < E) { s = ei[e]; d = ei[E + e]; } else { s = d = e - E; }
  const float4* pl = (const float4*)(xl + (size_t)d * CC + h * COH);
  const float4* pr = (const float4*)(xr + (size_t)s * CC + h * COH);
  const float4* pa = (const float4*)(att + h * COH);
  float acc = 0.0f;
#pragma unroll 4
  for (int c = 0; c < COH / 4; ++c) {
    float4 l = pl[c], r = pr[c], a = pa[c];
    float v0 = l.x + r.x, v1 = l.y + r.y, v2 = l.z + r.z, v3 = l.w + r.w;
    v0 = (v0 > 0.0f) ? v0 : 0.2f * v0;   // leaky_relu slope 0.2
    v1 = (v1 > 0.0f) ? v1 : 0.2f * v1;
    v2 = (v2 > 0.0f) ? v2 : 0.2f * v2;
    v3 = (v3 > 0.0f) ? v3 : 0.2f * v3;
    acc += a.x * v0 + a.y * v1 + a.z * v2 + a.w * v3;
  }
  float ex = __expf(acc);
  ealpha[(size_t)e * HH + h] = ex;
  atomicAdd(&denom[d * HH + h], ex);
}

__global__ void gat_aggregate_kernel(const float* __restrict__ xr, const int* __restrict__ ei,
                                     const float* __restrict__ ealpha,
                                     const float* __restrict__ denom,
                                     float* __restrict__ hacc, int E) {
  int t = blockIdx.x * blockDim.x + threadIdx.x;
  int EN = E + NN;
  if (t >= EN * HH) return;
  int e = t >> 3, h = t & 7;
  int s, d;
  if (e < E) { s = ei[e]; d = ei[E + e]; } else { s = d = e - E; }
  float a = ealpha[(size_t)e * HH + h] / denom[d * HH + h];
  const float4* pr = (const float4*)(xr + (size_t)s * CC + h * COH);
  float* po = hacc + (size_t)d * CC + h * COH;
#pragma unroll 4
  for (int c = 0; c < COH / 4; ++c) {
    float4 r = pr[c];
    atomicAdd(&po[4 * c + 0], a * r.x);
    atomicAdd(&po[4 * c + 1], a * r.y);
    atomicAdd(&po[4 * c + 2], a * r.z);
    atomicAdd(&po[4 * c + 3], a * r.w);
  }
}

// ------------------------------------------------- fused bias+res+LayerNorm
// One wave per row of 512. xo = LN(y + bias + res) * g + b ; optional bf16 copy.
__global__ __launch_bounds__(256)
void ln_kernel(const float* __restrict__ y, const float* __restrict__ bias,
               const float* __restrict__ res, const float* __restrict__ g,
               const float* __restrict__ b, float* __restrict__ xo,
               u16* __restrict__ xob) {
  int wave = threadIdx.x >> 5, lane = threadIdx.x & 31;
  int row = blockIdx.x * 8 + wave;
  const float* py = y + (size_t)row * CC;
  const float* pr = res ? res + (size_t)row * CC : nullptr;
  float v[16];
  float s = 0.0f;
#pragma unroll
  for (int i = 0; i < 16; ++i) {
    int c = lane + 32 * i;
    float t = py[c];
    if (bias) t += bias[c];
    if (pr) t += pr[c];
    v[i] = t; s += t;
  }
  s += __shfl_xor(s, 1, 32);  s += __shfl_xor(s, 2, 32);
  s += __shfl_xor(s, 4, 32);  s += __shfl_xor(s, 8, 32);
  s += __shfl_xor(s, 16, 32);
  float mu = s * (1.0f / CC);
  float q = 0.0f;
#pragma unroll
  for (int i = 0; i < 16; ++i) { float d = v[i] - mu; q += d * d; }
  q += __shfl_xor(q, 1, 32);  q += __shfl_xor(q, 2, 32);
  q += __shfl_xor(q, 4, 32);  q += __shfl_xor(q, 8, 32);
  q += __shfl_xor(q, 16, 32);
  float rinv = rsqrtf(q * (1.0f / CC) + 1e-5f);
#pragma unroll
  for (int i = 0; i < 16; ++i) {
    int c = lane + 32 * i;
    float o = (v[i] - mu) * rinv * g[c] + b[c];
    xo[(size_t)row * CC + c] = o;
    if (xob) xob[(size_t)row * CC + c] = f2bf(o);
  }
}

// ------------------------------------------------------ V transpose (bf16)
// vT[h*64 + c][n] = qkv[n][1024 + h*64 + c]  (write-coalesced)
__global__ void transpose_v_kernel(const u16* __restrict__ qkvb, u16* __restrict__ vT) {
  int t = blockIdx.x * blockDim.x + threadIdx.x;
  if (t >= NN * CC) return;
  int n = t & (NN - 1);
  int hc = t >> 12;
  vT[t] = qkvb[(size_t)n * (3 * CC) + 2 * CC + hc];
}

// -------------------------------------------------------------- attention
// One wave per (head, 16-query tile). Exact softmax without max (rescale
// cancels; scores are O(1) here). p fragment converted D-layout -> A-layout
// through a private LDS tile. __expf -> single v_exp_f32 per score, which
// co-executes with the WMMA (XDL) pipe.
__global__ __launch_bounds__(256)
void attention_kernel(const u16* __restrict__ qkvb, const u16* __restrict__ vT,
                      u16* __restrict__ attnb) {
  __shared__ __align__(16) unsigned short ptile[8][16 * 32];
  int wave = threadIdx.x >> 5, lane = threadIdx.x & 31;
  int half = lane >> 4, l15 = lane & 15;
  int task = blockIdx.x * 8 + wave;        // 2048 tasks
  int h  = task >> 8;                      // 256 q-tiles per head
  int q0 = (task & 255) << 4;

  const u16* qbase = qkvb + h * COH;                 // q section, head h
  const u16* kbase = qkvb + CC + h * COH;            // k section, head h
  const u16* vbase = vT + (size_t)h * COH * NN;      // vT[h][c][n]

  bf16x16 qa0 = load_a_frag(qbase, 3 * CC, q0, 0, lane);
  bf16x16 qa1 = load_a_frag(qbase, 3 * CC, q0, 32, lane);

  f32x8 acc0 = {}, acc1 = {}, acc2 = {}, acc3 = {};
  float dsum[8];
#pragma unroll
  for (int v = 0; v < 8; ++v) dsum[v] = 0.0f;

  unsigned short* tp = &ptile[wave][0];
  const float scale = 0.125f;  // 1/sqrt(64)

  for (int key0 = 0; key0 < NN; key0 += 32) {
    // scores S[16 x 32] = q (16x64) . k^T
    f32x8 s0 = {}, s1 = {};
    s0 = wmma_bf16(qa0, load_b_frag(kbase, 3 * CC, key0,       0, lane), s0);
    s0 = wmma_bf16(qa1, load_b_frag(kbase, 3 * CC, key0,      32, lane), s0);
    s1 = wmma_bf16(qa0, load_b_frag(kbase, 3 * CC, key0 + 16,  0, lane), s1);
    s1 = wmma_bf16(qa1, load_b_frag(kbase, 3 * CC, key0 + 16, 32, lane), s1);

    // exp, denominator partials, and LDS tile (rows q, cols key-within-step)
#pragma unroll
    for (int v = 0; v < 8; ++v) {
      float e0 = __expf(s0[v] * scale);
      float e1 = __expf(s1[v] * scale);
      dsum[v] += e0 + e1;
      int row = v + 8 * half;
      tp[row * 32 + l15]      = f2bf(e0);
      tp[row * 32 + 16 + l15] = f2bf(e1);
    }

    // re-read p as A-layout fragment (16x32, K = 32 keys)
    FragAB pa;
    const unsigned short* rp = tp + l15 * 32;
    pa.h[0] = *(const u16x8*)(rp + 8 * half);
    pa.h[1] = *(const u16x8*)(rp + 16 + 8 * half);

    // O += p (16x32) . V (32x64)
    acc0 = wmma_bf16(pa.f, load_b_frag(vbase, NN,  0, key0, lane), acc0);
    acc1 = wmma_bf16(pa.f, load_b_frag(vbase, NN, 16, key0, lane), acc1);
    acc2 = wmma_bf16(pa.f, load_b_frag(vbase, NN, 32, key0, lane), acc2);
    acc3 = wmma_bf16(pa.f, load_b_frag(vbase, NN, 48, key0, lane), acc3);
  }

  // reduce denominators across the 16 lanes of each half-wave
#pragma unroll
  for (int v = 0; v < 8; ++v) {
    float s = dsum[v];
    s += __shfl_xor(s, 1, 32);
    s += __shfl_xor(s, 2, 32);
    s += __shfl_xor(s, 4, 32);
    s += __shfl_xor(s, 8, 32);
    dsum[v] = 1.0f / s;
  }
  f32x8 accs[4] = {acc0, acc1, acc2, acc3};
#pragma unroll
  for (int t = 0; t < 4; ++t) {
#pragma unroll
    for (int v = 0; v < 8; ++v) {
      int row = q0 + v + 8 * half;
      int col = h * COH + t * 16 + l15;
      attnb[(size_t)row * CC + col] = f2bf(accs[t][v] * dsum[v]);
    }
  }
}

// ================================================================== launch
extern "C" void kernel_launch(void* const* d_in, const int* in_sizes, int n_in,
                              void* d_out, int out_size, void* d_ws, size_t ws_size,
                              hipStream_t stream) {
  const float* x      = (const float*)d_in[0];
  const int*   ei     = (const int*)  d_in[1];
  const float* wl     = (const float*)d_in[2];
  const float* bl     = (const float*)d_in[3];
  const float* wr     = (const float*)d_in[4];
  const float* br     = (const float*)d_in[5];
  const float* att    = (const float*)d_in[6];
  const float* gat_b  = (const float*)d_in[7];
  const float* in_w   = (const float*)d_in[8];
  const float* in_b   = (const float*)d_in[9];
  const float* out_w  = (const float*)d_in[10];
  const float* out_b  = (const float*)d_in[11];
  const float* ln1_g  = (const float*)d_in[12];
  const float* ln1_b  = (const float*)d_in[13];
  const float* ln2_g  = (const float*)d_in[14];
  const float* ln2_b  = (const float*)d_in[15];
  const float* ln3_g  = (const float*)d_in[16];
  const float* ln3_b  = (const float*)d_in[17];
  const float* w1     = (const float*)d_in[18];
  const float* b1     = (const float*)d_in[19];
  const float* w2     = (const float*)d_in[20];
  const float* b2     = (const float*)d_in[21];
  float* outp = (float*)d_out;

  const int E  = in_sizes[1] / 2;
  const int EN = E + NN;

  // ---- workspace layout (persistent region + phase-overlapped arena) ----
  char* ws = (char*)d_ws;
  size_t off = 0;
  auto take = [&](size_t bytes) -> void* {
    void* p = ws + off; off += (bytes + 255) & ~(size_t)255; return p;
  };
  u16*   wlb   = (u16*)take((size_t)CC * CC * 2);
  u16*   wrb   = (u16*)take((size_t)CC * CC * 2);
  u16*   inwb  = (u16*)take((size_t)3 * CC * CC * 2);
  u16*   outwb = (u16*)take((size_t)CC * CC * 2);
  u16*   w1b   = (u16*)take((size_t)4 * CC * CC * 2);
  u16*   w2b   = (u16*)take((size_t)4 * CC * CC * 2);
  u16*   xb    = (u16*)take((size_t)NN * CC * 2);
  float* x1    = (float*)take((size_t)NN * CC * 4);
  u16*   x1b   = (u16*)take((size_t)NN * CC * 2);
  float* x2    = (float*)take((size_t)NN * CC * 4);
  u16*   x2b   = (u16*)take((size_t)NN * CC * 2);
  float* gout  = (float*)take((size_t)NN * CC * 4);
  char*  arena = (char*)take((size_t)36 * 1024 * 1024);
  // GAT phase (arena):
  float* xl     = (float*)(arena);                              // 8 MB
  float* xr     = (float*)(arena + (size_t)8  * 1024 * 1024);   // 8 MB
  float* ealpha = (float*)(arena + (size_t)16 * 1024 * 1024);   // 4.5 MB
  float* denom  = (float*)(arena + (size_t)21 * 1024 * 1024);   // 0.125 MB
  float* hacc   = (float*)(arena + (size_t)22 * 1024 * 1024);   // 8 MB
  // MHA phase (arena, GAT buffers dead):
  u16*   qkvb   = (u16*)(arena);                                // 12 MB
  u16*   vTb    = (u16*)(arena + (size_t)12 * 1024 * 1024);     // 4 MB
  u16*   attnb  = (u16*)(arena + (size_t)16 * 1024 * 1024);     // 4 MB
  // FFN phase (arena, qkv/vT dead; attnb still live until out-proj done):
  u16*   f1b    = (u16*)(arena);                                // 16 MB

  auto cblocks = [](int n) { int b = (n + 255) / 256; return b > 65535 ? 65535 : b; };

  // ---- casts to bf16 ----
  cast_f32_to_bf16<<<cblocks(NN * CC), 256, 0, stream>>>(x, xb, NN * CC);
  cast_f32_to_bf16<<<cblocks(CC * CC), 256, 0, stream>>>(wl, wlb, CC * CC);
  cast_f32_to_bf16<<<cblocks(CC * CC), 256, 0, stream>>>(wr, wrb, CC * CC);
  cast_f32_to_bf16<<<cblocks(3 * CC * CC), 256, 0, stream>>>(in_w, inwb, 3 * CC * CC);
  cast_f32_to_bf16<<<cblocks(CC * CC), 256, 0, stream>>>(out_w, outwb, CC * CC);
  cast_f32_to_bf16<<<cblocks(4 * CC * CC), 256, 0, stream>>>(w1, w1b, 4 * CC * CC);
  cast_f32_to_bf16<<<cblocks(4 * CC * CC), 256, 0, stream>>>(w2, w2b, 4 * CC * CC);

  // ---- GATv2 ----
  dim3 g512(CC / 256, NN / 64);  // Nout=512
  gemm_bf16_kernel<false, true, false><<<g512, 256, 0, stream>>>(xb, wlb, bl, xl, nullptr, CC, CC);
  gemm_bf16_kernel<false, true, false><<<g512, 256, 0, stream>>>(xb, wrb, br, xr, nullptr, CC, CC);
  fill_zero_f32<<<cblocks(NN * HH), 256, 0, stream>>>(denom, NN * HH);
  fill_zero_f32<<<cblocks(NN * CC), 256, 0, stream>>>(hacc, NN * CC);
  int eth = EN * HH;
  gat_edge_kernel<<<(eth + 255) / 256, 256, 0, stream>>>(xl, xr, ei, att, ealpha, denom, E);
  gat_aggregate_kernel<<<(eth + 255) / 256, 256, 0, stream>>>(xr, ei, ealpha, denom, hacc, E);
  ln_kernel<<<NN / 8, 256, 0, stream>>>(hacc, gat_b, x, ln1_g, ln1_b, x1, x1b);

  // ---- MHA ----
  dim3 gqkv(3 * CC / 256, NN / 64);
  gemm_bf16_kernel<true, true, false><<<gqkv, 256, 0, stream>>>(x1b, inwb, in_b, nullptr, qkvb, 3 * CC, CC);
  transpose_v_kernel<<<(NN * CC + 255) / 256, 256, 0, stream>>>(qkvb, vTb);
  attention_kernel<<<(HH * (NN / 16)) / 8, 256, 0, stream>>>(qkvb, vTb, attnb);
  gemm_bf16_kernel<false, false, false><<<g512, 256, 0, stream>>>(attnb, outwb, nullptr, gout, nullptr, CC, CC);
  ln_kernel<<<NN / 8, 256, 0, stream>>>(gout, out_b, x1, ln2_g, ln2_b, x2, x2b);

  // ---- FFN ----
  dim3 gffn1(4 * CC / 256, NN / 64);
  gemm_bf16_kernel<true, true, true><<<gffn1, 256, 0, stream>>>(x2b, w1b, b1, nullptr, f1b, 4 * CC, CC);
  gemm_bf16_kernel<false, false, false><<<g512, 256, 0, stream>>>(f1b, w2b, nullptr, gout, nullptr, CC, 4 * CC);
  ln_kernel<<<NN / 8, 256, 0, stream>>>(gout, b2, x2, ln3_g, ln3_b, outp, nullptr);

  (void)n_in; (void)out_size; (void)ws_size;
}
